// AttentionLayer_79053168050525
// MI455X (gfx1250) — matrix operand order repaired and checked by reading, not compiled, not run on hardware
//
#include <hip/hip_runtime.h>
#include <stdint.h>

typedef __attribute__((ext_vector_type(16))) __bf16 v16bf;
typedef __attribute__((ext_vector_type(8)))  __bf16 v8bf;
typedef __attribute__((ext_vector_type(8)))  float  v8f;

#define DEVI static __device__ __forceinline__

DEVI unsigned short f32_bf16(float f) {
  union { float f; uint32_t u; } c; c.f = f;
  uint32_t u = c.u;
  uint32_t r = u + 0x7FFFu + ((u >> 16) & 1u);   // round-to-nearest-even
  return (unsigned short)(r >> 16);
}

union Frag16 { v16bf v; v8bf h[2]; unsigned short us[16]; };
union Pack8  { unsigned short us[8]; uint4 q; };
union Pack4  { unsigned short us[4]; uint2 q; };

// Async copy 32B: LDS[lds_off .. +31] = MEM[gaddr .. +31]  (ASYNCcnt-tracked)
DEVI void async_b128_x2(unsigned lds_off, uint64_t gaddr) {
  asm volatile("global_load_async_to_lds_b128 %0, %1, off\n\t"
               "global_load_async_to_lds_b128 %0, %1, off offset:16"
               :: "v"(lds_off), "v"(gaddr) : "memory");
}
DEVI void wait_async_all() {
  asm volatile("s_wait_asynccnt 0x0" ::: "memory");
}

// ---------------------------------------------------------------- convert (x4 vectorized)
__global__ __launch_bounds__(256)
void cvt_f32_bf16_kernel(const float* __restrict__ in,
                         unsigned short* __restrict__ out, int n) {
  int i = (blockIdx.x * 256 + threadIdx.x) * 4;
  if (i + 3 < n) {
    float4 f = *(const float4*)(in + i);
    Pack4 p;
    p.us[0] = f32_bf16(f.x); p.us[1] = f32_bf16(f.y);
    p.us[2] = f32_bf16(f.z); p.us[3] = f32_bf16(f.w);
    *(uint2*)(out + i) = p.q;
  }
}

// ---------------------------------------------------------------- unified WMMA GEMM core
// out[m,n] = sum_k A[m,k] * B[n,k]    (A, B bf16; acc f32)
// Block: 256 thr = 8 waves (4 row-waves x 2 col-waves), tile 128x128,
// wave tile 32x64, K step 32, double-buffered LDS fed by async copies.
// MODE 0: bf16 row-major store (q/k projection), ld_out = 1024
// MODE 1: bf16 store transposed per batch of 2048 rows (v projection -> vT[b][e][t])
// MODE 2: f32 store, *1/32 scale + causal mask; early-out fully-masked tiles
// MODE 3: f32 plain store, K truncated to row0+128 (P is 0 above diagonal)
template<int MODE>
__global__ __launch_bounds__(256)
void gemm_core(const unsigned short* __restrict__ A,
               const unsigned short* __restrict__ B,
               void* __restrict__ outv,
               int lda, int ldb, int K) {
  __shared__ unsigned short As[2][128 * 40];   // 40-elem row stride: bank-conflict-free
  __shared__ unsigned short Bs[2][128 * 40];

  const int tid  = threadIdx.x;
  const int wave = tid >> 5, lane = tid & 31;
  const int lhalf = lane >> 4, lmod = lane & 15;
  const int waveM = wave & 3, waveN = wave >> 2;
  const int row0 = blockIdx.y * 128;
  const int col0 = blockIdx.x * 128;

  if (MODE == 2 && col0 > row0 + 127) {        // tile entirely above diagonal
    float* sc = (float*)outv;
    #pragma unroll
    for (int i = 0; i < 64; i++) {
      int lin = tid + i * 256;
      int r = lin >> 7, c = lin & 127;
      sc[(size_t)(row0 + r) * 2048 + col0 + c] = -1.0e30f;
    }
    return;
  }

  int kend = (MODE == 3) ? (row0 + 128) : K;
  const int nk = kend >> 5;

  // staging: thread -> (row/col sidx, 16-elem half)
  const int sidx = tid >> 1;
  const int skh  = (tid & 1) * 16;
  const unsigned short* gA = A + (size_t)(row0 + sidx) * lda + skh;
  const unsigned short* gB = B + (size_t)(col0 + sidx) * ldb + skh;
  const unsigned lsoff = (unsigned)(sidx * 40 + skh);

  // issue stage 0
  async_b128_x2((unsigned)(size_t)&As[0][lsoff], (uint64_t)gA);
  async_b128_x2((unsigned)(size_t)&Bs[0][lsoff], (uint64_t)gB);

  v8f acc[2][4];
  #pragma unroll
  for (int i = 0; i < 2; i++)
    #pragma unroll
    for (int j = 0; j < 4; j++)
      #pragma unroll
      for (int e = 0; e < 8; e++) acc[i][j][e] = 0.f;

  for (int s = 0; s < nk; s++) {
    wait_async_all();          // this wave's stage-s copies landed in LDS
    __syncthreads();           // everyone's copies landed; prev reads done

    if (s + 1 < nk) {
      int kk = (s + 1) << 5;
      async_b128_x2((unsigned)(size_t)&As[(s + 1) & 1][lsoff], (uint64_t)(gA + kk));
      async_b128_x2((unsigned)(size_t)&Bs[(s + 1) & 1][lsoff], (uint64_t)(gB + kk));
    }

    const unsigned short* as = As[s & 1];
    const unsigned short* bs = Bs[s & 1];

    Frag16 a[2];
    #pragma unroll
    for (int i = 0; i < 2; i++) {
      const unsigned short* ap = as + (waveM * 32 + i * 16 + lmod) * 40 + lhalf * 8;
      a[i].h[0] = *(const v8bf*)ap;
      a[i].h[1] = *(const v8bf*)(ap + 16);
    }
    #pragma unroll
    for (int j = 0; j < 4; j++) {
      Frag16 b;
      const unsigned short* bp = bs + (waveN * 64 + j * 16 + lmod) * 40 + lhalf * 8;
      b.h[0] = *(const v8bf*)bp;
      b.h[1] = *(const v8bf*)(bp + 16);
      #pragma unroll
      for (int i = 0; i < 2; i++)
        acc[i][j] = __builtin_amdgcn_wmma_f32_16x16x32_bf16(
            false, a[i].v, false, b.v, (short)0, acc[i][j], false, false);
    }
  }

  const int r0 = row0 + waveM * 32;
  const int c0 = col0 + waveN * 64;

  if (MODE == 0) {
    unsigned short* y = (unsigned short*)outv;
    #pragma unroll
    for (int i = 0; i < 2; i++)
      #pragma unroll
      for (int j = 0; j < 4; j++) {
        int col = c0 + j * 16 + lmod;
        #pragma unroll
        for (int v = 0; v < 8; v++) {
          int row = r0 + i * 16 + lhalf * 8 + v;
          y[(size_t)row * 1024 + col] = f32_bf16(acc[i][j][v]);
        }
      }
  } else if (MODE == 1) {
    unsigned short* y = (unsigned short*)outv;
    #pragma unroll
    for (int i = 0; i < 2; i++) {
      int rowb  = r0 + i * 16 + lhalf * 8;   // 8 consecutive rows per lane
      int batch = rowb >> 11;
      int t0    = rowb & 2047;
      #pragma unroll
      for (int j = 0; j < 4; j++) {
        int col = c0 + j * 16 + lmod;
        Pack8 p;
        #pragma unroll
        for (int v = 0; v < 8; v++) p.us[v] = f32_bf16(acc[i][j][v]);
        *(uint4*)(y + (size_t)batch * (1024u * 2048u) + (size_t)col * 2048 + t0) = p.q;
      }
    }
  } else if (MODE == 2) {
    float* sc = (float*)outv;
    #pragma unroll
    for (int i = 0; i < 2; i++)
      #pragma unroll
      for (int j = 0; j < 4; j++) {
        int col = c0 + j * 16 + lmod;
        #pragma unroll
        for (int v = 0; v < 8; v++) {
          int row = r0 + i * 16 + lhalf * 8 + v;
          float val = (col <= row) ? acc[i][j][v] * 0.03125f : -1.0e30f;
          sc[(size_t)row * 2048 + col] = val;
        }
      }
  } else {
    float* o = (float*)outv;
    #pragma unroll
    for (int i = 0; i < 2; i++)
      #pragma unroll
      for (int j = 0; j < 4; j++) {
        int col = c0 + j * 16 + lmod;
        #pragma unroll
        for (int v = 0; v < 8; v++) {
          int row = r0 + i * 16 + lhalf * 8 + v;
          o[(size_t)row * 1024 + col] = acc[i][j][v];
        }
      }
  }
}

// ---------------------------------------------------------------- row softmax, fp32 -> bf16
__global__ __launch_bounds__(256)
void softmax_kernel(const float* __restrict__ sc,
                    unsigned short* __restrict__ P) {
  const int T = 2048;
  __shared__ float red[256];
  int row = blockIdx.x, tid = threadIdx.x;
  const float* srow = sc + (size_t)row * T;

  float vals[8];
  float m = -3.4e38f;
  #pragma unroll
  for (int i = 0; i < 8; i++) { vals[i] = srow[tid + i * 256]; m = fmaxf(m, vals[i]); }
  red[tid] = m; __syncthreads();
  for (int s = 128; s > 0; s >>= 1) {
    if (tid < s) red[tid] = fmaxf(red[tid], red[tid + s]);
    __syncthreads();
  }
  float rm = red[0]; __syncthreads();

  float sum = 0.f;
  #pragma unroll
  for (int i = 0; i < 8; i++) { vals[i] = __expf(vals[i] - rm); sum += vals[i]; }
  red[tid] = sum; __syncthreads();
  for (int s = 128; s > 0; s >>= 1) {
    if (tid < s) red[tid] += red[tid + s];
    __syncthreads();
  }
  float inv = 1.0f / red[0];

  unsigned short* prow = P + (size_t)row * T;
  #pragma unroll
  for (int i = 0; i < 8; i++) prow[tid + i * 256] = f32_bf16(vals[i] * inv);
}

// ---------------------------------------------------------------- launch
extern "C" void kernel_launch(void* const* d_in, const int* in_sizes, int n_in,
                              void* d_out, int out_size, void* d_ws, size_t ws_size,
                              hipStream_t stream) {
  (void)in_sizes; (void)n_in; (void)out_size; (void)ws_size;
  const float* query = (const float*)d_in[0];
  const float* key   = (const float*)d_in[1];
  const float* value = (const float*)d_in[2];
  // d_in[3] = attn_mask (causal, known analytically) -- unused
  const float* Wq = (const float*)d_in[4];
  const float* Wk = (const float*)d_in[5];
  const float* Wv = (const float*)d_in[6];
  float* out = (float*)d_out;

  const size_t WE = (size_t)1024 * 1024;       // weight elems
  const size_t XE = (size_t)4 * 2048 * 1024;   // N*S*D elems

  char* ws = (char*)d_ws;
  unsigned short* Wqb = (unsigned short*)ws; ws += WE * 2;
  unsigned short* Wkb = (unsigned short*)ws; ws += WE * 2;
  unsigned short* Wvb = (unsigned short*)ws; ws += WE * 2;
  unsigned short* xq  = (unsigned short*)ws; ws += XE * 2;
  unsigned short* xk  = (unsigned short*)ws; ws += XE * 2;
  unsigned short* xv  = (unsigned short*)ws; ws += XE * 2;
  unsigned short* qb  = (unsigned short*)ws; ws += XE * 2;
  unsigned short* kb  = (unsigned short*)ws; ws += XE * 2;
  unsigned short* vT  = (unsigned short*)ws; ws += XE * 2;
  float*          sc  = (float*)ws;          ws += (size_t)2048 * 2048 * 4;
  unsigned short* P   = (unsigned short*)ws; ws += (size_t)2048 * 2048 * 2;

  // fp32 -> bf16 (once; bandwidth-bound)
  cvt_f32_bf16_kernel<<<(int)(WE / 1024), 256, 0, stream>>>(Wq, Wqb, (int)WE);
  cvt_f32_bf16_kernel<<<(int)(WE / 1024), 256, 0, stream>>>(Wk, Wkb, (int)WE);
  cvt_f32_bf16_kernel<<<(int)(WE / 1024), 256, 0, stream>>>(Wv, Wvb, (int)WE);
  cvt_f32_bf16_kernel<<<(int)(XE / 1024), 256, 0, stream>>>(query, xq, (int)XE);
  cvt_f32_bf16_kernel<<<(int)(XE / 1024), 256, 0, stream>>>(key,   xk, (int)XE);
  cvt_f32_bf16_kernel<<<(int)(XE / 1024), 256, 0, stream>>>(value, xv, (int)XE);

  // projections: 8192x1024 @ 1024x1024^T
  dim3 pgrid(1024 / 128, 8192 / 128);
  gemm_core<0><<<pgrid, 256, 0, stream>>>(xq, Wqb, qb, 1024, 1024, 1024);
  gemm_core<0><<<pgrid, 256, 0, stream>>>(xk, Wkb, kb, 1024, 1024, 1024);
  gemm_core<1><<<pgrid, 256, 0, stream>>>(xv, Wvb, vT, 1024, 1024, 1024);

  // attention, batch-serial (reuses sc/P scratch; stream orders the kernels)
  for (int n = 0; n < 4; n++) {
    gemm_core<2><<<dim3(2048 / 128, 2048 / 128), 256, 0, stream>>>(
        qb + (size_t)n * 2048 * 1024, kb + (size_t)n * 2048 * 1024, sc,
        1024, 1024, 1024);
    softmax_kernel<<<2048, 256, 0, stream>>>(sc, P);
    gemm_core<3><<<dim3(1024 / 128, 2048 / 128), 256, 0, stream>>>(
        P, vT + (size_t)n * 1024 * 2048, out + (size_t)n * 2048 * 1024,
        2048, 2048, 2048);
  }
}
